// GPSConv_8976481649041
// MI455X (gfx1250) — compile-verified, matrix operand order checked by reading, not run on hardware
//
#include <hip/hip_runtime.h>
#include <stdint.h>

// ---------------------------------------------------------------------------
// Shapes (fixed by the reference)
// ---------------------------------------------------------------------------
#define Bn    8
#define Nn    1024
#define Cc    256
#define Hh    8
#define DKk   32
#define Ee    131072
#define BNn   (Bn * Nn)          // 8192
#define EPSf  1e-5f

typedef __attribute__((ext_vector_type(16))) _Float16 v16h;
typedef __attribute__((ext_vector_type(8)))  float    v8f;

union FragH { v16h v; uint32_t u[8]; _Float16 h[16]; };
union FragF { v8f  v; float f[8]; };

typedef __attribute__((address_space(3))) _Float16 lds_f16_t;

// ---------------------------------------------------------------------------
// CDNA5 async global->LDS copy (ASYNCcnt-tracked, bypasses VGPRs)
// ---------------------------------------------------------------------------
__device__ __forceinline__ void async_copy_b128(void* lds_dst, const void* gsrc) {
  uint32_t loff = (uint32_t)(uintptr_t)(lds_f16_t*)lds_dst;   // LDS byte offset
  unsigned long long ga = (unsigned long long)(uintptr_t)gsrc;
  asm volatile("global_load_async_to_lds_b128 %0, %1, off"
               :: "v"(loff), "v"(ga) : "memory");
}
__device__ __forceinline__ void wait_async() {
  asm volatile("s_wait_asynccnt 0x0" ::: "memory");
}

// ---------------------------------------------------------------------------
// WMMA helpers
// ---------------------------------------------------------------------------
__device__ __forceinline__ v8f wmma_f16(v16h a, v16h b, v8f c) {
  return __builtin_amdgcn_wmma_f32_16x16x32_f16(false, a, false, b, (short)0, c,
                                                false, false);
}

// Build a 16x32 f16 WMMA operand fragment from LDS, row-major with `stride`
// (f16 elems, even). Matches the ISA 16-bit A/B operand layout:
//  lanes 0-15  : row = lane,    K = {0..7, 16..23}
//  lanes 16-31 : row = lane-16, K = {8..15, 24..31}
__device__ __forceinline__ v16h ld_frag(const _Float16* base, int stride) {
  int lane = threadIdx.x & 31;
  int lr   = lane & 15;
  int hs   = lane >> 4;
  const _Float16* p = base + lr * stride + hs * 8;
  FragH f;
#pragma unroll
  for (int v = 0; v < 4; ++v)
    f.u[v] = *(const uint32_t*)(p + 2 * v);
#pragma unroll
  for (int v = 0; v < 4; ++v)
    f.u[4 + v] = *(const uint32_t*)(p + 16 + 2 * v);
  return f.v;
}

__device__ __forceinline__ float rmax16(float v) {
#pragma unroll
  for (int o = 8; o > 0; o >>= 1) v = fmaxf(v, __shfl_xor(v, o, 32));
  return v;
}
__device__ __forceinline__ float rsum16(float v) {
#pragma unroll
  for (int o = 8; o > 0; o >>= 1) v += __shfl_xor(v, o, 32);
  return v;
}

// ---------------------------------------------------------------------------
// Generic tiled WMMA GEMM:  Y[M,N] = act(X[M,K] * W[K,N] + bias)
// X is f16 [M,K]; Wt is the PRE-TRANSPOSED weight, f16 [N,K].
// Double-buffered LDS, async global->LDS staging. 256 thr = 8 waves,
// block tile 64x64, wave tile 16x32, K step 32.
// ---------------------------------------------------------------------------
#define LDT 40

__global__ __launch_bounds__(256) void gemm_wmma(
    const _Float16* __restrict__ A, const _Float16* __restrict__ Wt,
    const float* __restrict__ bias, float* __restrict__ outF,
    _Float16* __restrict__ outH, int M, int N, int K, int relu) {
  __shared__ _Float16 As[2][64 * LDT];    // As[r][k]
  __shared__ _Float16 Bs[2][64 * LDT];    // Bs[n][k]

  const int tid  = threadIdx.x;
  const int lane = tid & 31;
  const int wave = tid >> 5;
  const int lr   = lane & 15;
  const int hs   = lane >> 4;
  const int wrow = wave & 3;
  const int wcol = wave >> 2;
  const int row0 = blockIdx.y * 64;
  const int col0 = blockIdx.x * 64;

  // per-thread staging slot: 8 contiguous f16 (one b128)
  const int sr = tid >> 2;                 // 0..63
  const int sk = (tid & 3) * 8;            // 0,8,16,24
  const _Float16* ga = A  + (size_t)(row0 + sr) * K + sk;
  const _Float16* gb = Wt + (size_t)(col0 + sr) * K + sk;

  async_copy_b128(&As[0][sr * LDT + sk], ga);
  async_copy_b128(&Bs[0][sr * LDT + sk], gb);

  v8f acc0 = {}, acc1 = {};
  const int nk = K >> 5;
  for (int i = 0; i < nk; ++i) {
    wait_async();
    __syncthreads();
    if (i + 1 < nk) {   // prefetch next K tile into the other buffer
      int nb = (i + 1) & 1, k0 = (i + 1) << 5;
      async_copy_b128(&As[nb][sr * LDT + sk], ga + k0);
      async_copy_b128(&Bs[nb][sr * LDT + sk], gb + k0);
    }
    const int buf = i & 1;
    v16h a  = ld_frag(&As[buf][(wrow * 16) * LDT], LDT);
    v16h b0 = ld_frag(&Bs[buf][(wcol * 32) * LDT], LDT);
    v16h b1 = ld_frag(&Bs[buf][(wcol * 32 + 16) * LDT], LDT);
    acc0 = wmma_f16(a, b0, acc0);
    acc1 = wmma_f16(a, b1, acc1);
  }

  FragF c0, c1; c0.v = acc0; c1.v = acc1;
  const int rbase = row0 + wrow * 16 + hs * 8;
  const int cb0   = col0 + wcol * 32 + lr;
  const int cb1   = cb0 + 16;
  const float bi0 = bias ? bias[cb0] : 0.f;
  const float bi1 = bias ? bias[cb1] : 0.f;
#pragma unroll
  for (int j = 0; j < 8; ++j) {
    int r = rbase + j;
    float v0 = c0.f[j] + bi0;
    float v1 = c1.f[j] + bi1;
    if (relu) { v0 = fmaxf(v0, 0.f); v1 = fmaxf(v1, 0.f); }
    if (outF) {
      outF[(size_t)r * N + cb0] = v0;
      outF[(size_t)r * N + cb1] = v1;
    }
    if (outH) {
      outH[(size_t)r * N + cb0] = (_Float16)v0;
      outH[(size_t)r * N + cb1] = (_Float16)v1;
    }
  }
}

// ---------------------------------------------------------------------------
// Flash-style sph-modulated attention. One block = one (b,h) x 64-row strip.
// 128 threads = 4 waves; wave w owns rows [16w,16w+16).
// ---------------------------------------------------------------------------
__global__ __launch_bounds__(128) void attn_wmma(
    const _Float16* __restrict__ qh, const _Float16* __restrict__ kh,
    const _Float16* __restrict__ vh, const float* __restrict__ sph,
    _Float16* __restrict__ oh) {
  __shared__ _Float16 Qs[64 * 40];        // Qs[r][d]
  __shared__ _Float16 Ks[64 * 40];        // Ks[c][d]  (K^T staging, B operand)
  __shared__ _Float16 VsT[32 * 72];       // VsT[d][c] (V^T staging, B operand)
  __shared__ _Float16 Ps[4][16 * 72];     // per-wave P strip (A operand)

  const int bid  = blockIdx.x;
  const int b    = bid >> 7;              // / (H * 16)
  const int h    = (bid >> 4) & 7;
  const int row0 = (bid & 15) * 64;

  const int tid  = threadIdx.x;
  const int lane = tid & 31;
  const int wave = tid >> 5;
  const int lr   = lane & 15;
  const int hs   = lane >> 4;

  const int r2   = tid & 63;              // staging row
  const int off2 = (tid >> 6) * 16;       // staging col (0 or 16)

  {   // Q strip: 64 x 32, async global->LDS
    const _Float16* src = qh + (size_t)(b * Nn + row0 + r2) * Cc + h * 32 + off2;
    async_copy_b128(&Qs[r2 * 40 + off2], src);
    async_copy_b128(&Qs[r2 * 40 + off2 + 8], src + 8);
  }
  wait_async();
  __syncthreads();

  v16h aQ = ld_frag(&Qs[(wave * 16) * 40], 40);

  float rm[8], rl[8];
#pragma unroll
  for (int j = 0; j < 8; ++j) { rm[j] = -1e30f; rl[j] = 0.f; }
  v8f o0 = {}, o1 = {};
  const float scale = 0.17677669529663689f;   // 1/sqrt(32)

  for (int c0 = 0; c0 < Nn; c0 += 64) {
    {   // async K tile; V transposed via registers (overlaps with async K)
      const _Float16* ks = kh + (size_t)(b * Nn + c0 + r2) * Cc + h * 32 + off2;
      async_copy_b128(&Ks[r2 * 40 + off2], ks);
      async_copy_b128(&Ks[r2 * 40 + off2 + 8], ks + 8);
      const _Float16* vs = vh + (size_t)(b * Nn + c0 + r2) * Cc + h * 32 + off2;
      _Float16 tmp[16];
      *(uint4*)&tmp[0] = *(const uint4*)vs;
      *(uint4*)&tmp[8] = *(const uint4*)(vs + 8);
#pragma unroll
      for (int i = 0; i < 16; ++i) VsT[(off2 + i) * 72 + r2] = tmp[i];
    }
    wait_async();
    __syncthreads();

    // S = Q * K^T : 4 tiles of 16x16, one WMMA each (K-dim = DK = 32)
    FragF S[4];
#pragma unroll
    for (int t = 0; t < 4; ++t) {
      v16h bK = ld_frag(&Ks[(t * 16) * 40], 40);
      v8f z = {};
      S[t].v = wmma_f16(aQ, bK, z);
    }

    // scale + structural modulation by sph[b, row, col]
    const int rbase = row0 + wave * 16 + hs * 8;
#pragma unroll
    for (int t = 0; t < 4; ++t) {
#pragma unroll
      for (int j = 0; j < 8; ++j) {
        int row = rbase + j;
        int col = c0 + t * 16 + lr;
        float m = sph[(size_t)b * Nn * Nn + (size_t)row * Nn + col];
        S[t].f[j] = S[t].f[j] * scale * m;
      }
    }

    // online softmax (row stats via 16-lane shuffles)
    float alpha[8], bsum[8];
#pragma unroll
    for (int j = 0; j < 8; ++j) {
      float mx = fmaxf(fmaxf(S[0].f[j], S[1].f[j]), fmaxf(S[2].f[j], S[3].f[j]));
      mx = rmax16(mx);
      float mnew = fmaxf(rm[j], mx);
      alpha[j] = __expf(rm[j] - mnew);
      rm[j] = mnew;
      bsum[j] = 0.f;
    }
#pragma unroll
    for (int t = 0; t < 4; ++t)
#pragma unroll
      for (int j = 0; j < 8; ++j) {
        float p = __expf(S[t].f[j] - rm[j]);
        S[t].f[j] = p;
        bsum[j] += p;
      }
#pragma unroll
    for (int j = 0; j < 8; ++j) rl[j] = rl[j] * alpha[j] + rsum16(bsum[j]);

    // rescale running O, stage P into per-wave LDS strip (D-layout -> A-layout)
    FragF O0, O1; O0.v = o0; O1.v = o1;
#pragma unroll
    for (int j = 0; j < 8; ++j) { O0.f[j] *= alpha[j]; O1.f[j] *= alpha[j]; }
#pragma unroll
    for (int t = 0; t < 4; ++t)
#pragma unroll
      for (int j = 0; j < 8; ++j)
        Ps[wave][(j + hs * 8) * 72 + t * 16 + lr] = (_Float16)S[t].f[j];
    // same-wave LDS ops are in-order; compiler inserts dscnt waits.

    // O += P * V  (P: 16x64 -> two K=32 A-frags; V: 64x32 -> 2 d-tiles)
    v16h aP0  = ld_frag(&Ps[wave][0], 72);
    v16h aP1  = ld_frag(&Ps[wave][32], 72);
    v16h bV00 = ld_frag(&VsT[0 * 72 + 0], 72);
    v16h bV01 = ld_frag(&VsT[0 * 72 + 32], 72);
    v16h bV10 = ld_frag(&VsT[16 * 72 + 0], 72);
    v16h bV11 = ld_frag(&VsT[16 * 72 + 32], 72);
    o0 = wmma_f16(aP0, bV00, O0.v);
    o0 = wmma_f16(aP1, bV01, o0);
    o1 = wmma_f16(aP0, bV10, O1.v);
    o1 = wmma_f16(aP1, bV11, o1);
    __syncthreads();
  }

  // normalize and store O (f16) into [BN, C] with head offset
  FragF O0, O1; O0.v = o0; O1.v = o1;
  const int rbase = row0 + wave * 16 + hs * 8;
#pragma unroll
  for (int j = 0; j < 8; ++j) {
    float inv = 1.f / rl[j];
    size_t r = (size_t)(b * Nn + rbase + j);
    oh[r * Cc + h * 32 + lr]      = (_Float16)(O0.f[j] * inv);
    oh[r * Cc + h * 32 + 16 + lr] = (_Float16)(O1.f[j] * inv);
  }
}

// ---------------------------------------------------------------------------
// Edge scatter: agg[dst] += msg[src]  (4 channels per thread, f32 atomics)
// ---------------------------------------------------------------------------
__global__ void edge_scatter(const long long* __restrict__ src,
                             const long long* __restrict__ dst,
                             const float* __restrict__ msg,
                             float* __restrict__ agg) {
  int i = blockIdx.x * blockDim.x + threadIdx.x;
  if (i >= Ee * (Cc / 4)) return;
  int e = i / (Cc / 4);
  int c = (i % (Cc / 4)) * 4;
  long long s = src[e], d = dst[e];
  const float* m = msg + (size_t)s * Cc + c;
  float* a = agg + (size_t)d * Cc + c;
  atomicAdd(a + 0, m[0]);
  atomicAdd(a + 1, m[1]);
  atomicAdd(a + 2, m[2]);
  atomicAdd(a + 3, m[3]);
}

// ---------------------------------------------------------------------------
// Column mean / rsqrt(var+eps): one block per feature column
// ---------------------------------------------------------------------------
__global__ __launch_bounds__(256) void colstats(const float* __restrict__ X,
                                                int M, int N,
                                                float* __restrict__ mean,
                                                float* __restrict__ inv) {
  int c = blockIdx.x;
  float s = 0.f, sq = 0.f;
  for (int r = threadIdx.x; r < M; r += 256) {
    float v = X[(size_t)r * N + c];
    s += v; sq += v * v;
  }
  __shared__ float ss[256], qq[256];
  ss[threadIdx.x] = s; qq[threadIdx.x] = sq;
  __syncthreads();
  for (int o = 128; o > 0; o >>= 1) {
    if (threadIdx.x < o) {
      ss[threadIdx.x] += ss[threadIdx.x + o];
      qq[threadIdx.x] += qq[threadIdx.x + o];
    }
    __syncthreads();
  }
  if (threadIdx.x == 0) {
    float m = ss[0] / M;
    float v = qq[0] / M - m * m;
    mean[c] = m;
    inv[c]  = rsqrtf(v + EPSf);
  }
}

// ---------------------------------------------------------------------------
// Elementwise kernels
// ---------------------------------------------------------------------------
__global__ void cvt_f32_f16(const float* __restrict__ a, _Float16* __restrict__ o,
                            int n) {
  int i = blockIdx.x * blockDim.x + threadIdx.x;
  if (i < n) o[i] = (_Float16)a[i];
}

// transpose + convert: W [K,N] f32  ->  Wt [N,K] f16
__global__ void cvt_tr_f16(const float* __restrict__ w, _Float16* __restrict__ wt,
                           int K, int N) {
  int i = blockIdx.x * blockDim.x + threadIdx.x;
  if (i >= K * N) return;
  int k = i / N, n = i % N;
  wt[(size_t)n * K + k] = (_Float16)w[i];
}

__global__ void add3_f32(float* __restrict__ o, const float* __restrict__ a,
                         const float* __restrict__ b, const float* __restrict__ c,
                         int n) {
  int i = blockIdx.x * blockDim.x + threadIdx.x;
  if (i < n) o[i] = a[i] + b[i] + c[i];
}

__global__ void add2_f32(float* __restrict__ o, const float* __restrict__ a,
                         const float* __restrict__ b, int n) {
  int i = blockIdx.x * blockDim.x + threadIdx.x;
  if (i < n) o[i] = a[i] + b[i];
}

// s = bn1(t1) + bn2(t2), also emit f16 copy for the MLP GEMM
__global__ void bn2_sum(const float* __restrict__ t1, const float* __restrict__ t2,
                        const float* __restrict__ m1, const float* __restrict__ i1,
                        const float* __restrict__ g1, const float* __restrict__ b1,
                        const float* __restrict__ m2, const float* __restrict__ i2,
                        const float* __restrict__ g2, const float* __restrict__ b2,
                        float* __restrict__ s, _Float16* __restrict__ sh, int n) {
  int i = blockIdx.x * blockDim.x + threadIdx.x;
  if (i >= n) return;
  int c = i & (Cc - 1);
  float a = (t1[i] - m1[c]) * i1[c] * g1[c] + b1[c];
  float b = (t2[i] - m2[c]) * i2[c] * g2[c] + b2[c];
  float v = a + b;
  s[i]  = v;
  sh[i] = (_Float16)v;
}

__global__ void bn_apply(const float* __restrict__ t, const float* __restrict__ m,
                         const float* __restrict__ iv, const float* __restrict__ g,
                         const float* __restrict__ b, float* __restrict__ o, int n) {
  int i = blockIdx.x * blockDim.x + threadIdx.x;
  if (i >= n) return;
  int c = i & (Cc - 1);
  o[i] = (t[i] - m[c]) * iv[c] * g[c] + b[c];
}

// ---------------------------------------------------------------------------
// Host orchestration
// ---------------------------------------------------------------------------
extern "C" void kernel_launch(void* const* d_in, const int* in_sizes, int n_in,
                              void* d_out, int out_size, void* d_ws, size_t ws_size,
                              hipStream_t stream) {
  (void)in_sizes; (void)n_in; (void)out_size; (void)ws_size;

  const float*     x    = (const float*)d_in[0];
  const long long* ei   = (const long long*)d_in[1];       // int64 [2,E]
  const float*     sph  = (const float*)d_in[2];
  const float*     Wr   = (const float*)d_in[3];
  const float*     Wn   = (const float*)d_in[4];
  const float*     Wq   = (const float*)d_in[5];
  const float*     bq   = (const float*)d_in[6];
  const float*     Wk   = (const float*)d_in[7];
  const float*     bk   = (const float*)d_in[8];
  const float*     Wv   = (const float*)d_in[9];
  const float*     bv   = (const float*)d_in[10];
  const float*     Wo   = (const float*)d_in[11];
  const float*     bo   = (const float*)d_in[12];
  const float*     W1   = (const float*)d_in[13];
  const float*     b1   = (const float*)d_in[14];
  const float*     W2   = (const float*)d_in[15];
  const float*     b2   = (const float*)d_in[16];
  const float*     g1   = (const float*)d_in[17];
  const float*     be1  = (const float*)d_in[18];
  const float*     g2   = (const float*)d_in[19];
  const float*     be2  = (const float*)d_in[20];
  const float*     g3   = (const float*)d_in[21];
  const float*     be3  = (const float*)d_in[22];
  float*           out  = (float*)d_out;

  char* ws = (char*)d_ws;
  _Float16* xh   = (_Float16*)(ws + (0ull  << 20));  // 4 MiB
  _Float16* wh   = (_Float16*)(ws + (4ull  << 20));  // 2 MiB (all weights^T, f16)
  _Float16* qh   = (_Float16*)(ws + (6ull  << 20));  // 4 MiB
  _Float16* kh   = (_Float16*)(ws + (10ull << 20));  // 4 MiB
  _Float16* vh   = (_Float16*)(ws + (14ull << 20));  // 4 MiB
  _Float16* oh   = (_Float16*)(ws + (18ull << 20));  // 4 MiB attn output
  float*    bufA = (float*)(ws + (22ull << 20));     // 8 MiB  t1
  float*    bufB = (float*)(ws + (30ull << 20));     // 8 MiB  agg
  float*    bufC = (float*)(ws + (38ull << 20));     // 8 MiB  xWn then t2
  float*    sbuf = (float*)(ws + (46ull << 20));     // 8 MiB  s
  _Float16* sh   = (_Float16*)(ws + (54ull << 20));  // 4 MiB  s (f16)
  _Float16* uh   = (_Float16*)(ws + (58ull << 20));  // 8 MiB  relu hidden f16
  float*    bufD = (float*)(ws + (66ull << 20));     // 8 MiB  mlp/t3
  float*    st   = (float*)(ws + (74ull << 20));     // stats

  // transposed f16 weights: Wt[n][k]
  _Float16* WnT = wh + 0;
  _Float16* WrT = wh + 65536;
  _Float16* WqT = wh + 131072;
  _Float16* WkT = wh + 196608;
  _Float16* WvT = wh + 262144;
  _Float16* WoT = wh + 327680;
  _Float16* W1T = wh + 393216;   // [512,256]
  _Float16* W2T = wh + 524288;   // [256,512]

  float *m1 = st, *i1 = st + 256, *m2 = st + 512, *i2 = st + 768,
        *m3 = st + 1024, *i3 = st + 1280;

  const int NE = BNn * Cc;                 // 2,097,152
  const dim3 eb(256);
  const dim3 eg((NE + 255) / 256);

  // 0) f16 conversions (weights transposed once; reused by every tile)
  cvt_f32_f16<<<eg, eb, 0, stream>>>(x, xh, NE);
  cvt_tr_f16<<<dim3(256), eb, 0, stream>>>(Wn, WnT, Cc, Cc);
  cvt_tr_f16<<<dim3(256), eb, 0, stream>>>(Wr, WrT, Cc, Cc);
  cvt_tr_f16<<<dim3(256), eb, 0, stream>>>(Wq, WqT, Cc, Cc);
  cvt_tr_f16<<<dim3(256), eb, 0, stream>>>(Wk, WkT, Cc, Cc);
  cvt_tr_f16<<<dim3(256), eb, 0, stream>>>(Wv, WvT, Cc, Cc);
  cvt_tr_f16<<<dim3(256), eb, 0, stream>>>(Wo, WoT, Cc, Cc);
  cvt_tr_f16<<<dim3(512), eb, 0, stream>>>(W1, W1T, Cc, 512);
  cvt_tr_f16<<<dim3(512), eb, 0, stream>>>(W2, W2T, 512, Cc);

  const dim3 gb(256);
  const dim3 g256(Cc / 64, BNn / 64);      // (4, 128)
  const dim3 g512(512 / 64, BNn / 64);     // (8, 128)

  // 1) local branch: msg = x@Wn ; agg = segment_sum ; t1 = x@Wr + agg + x
  gemm_wmma<<<g256, gb, 0, stream>>>(xh, WnT, nullptr, bufC, nullptr,
                                     BNn, Cc, Cc, 0);
  hipMemsetAsync(bufB, 0, (size_t)NE * sizeof(float), stream);
  edge_scatter<<<dim3((Ee * (Cc / 4) + 255) / 256), eb, 0, stream>>>(
      ei, ei + Ee, bufC, bufB);
  gemm_wmma<<<g256, gb, 0, stream>>>(xh, WrT, nullptr, bufA, nullptr,
                                     BNn, Cc, Cc, 0);
  add3_f32<<<eg, eb, 0, stream>>>(bufA, bufA, bufB, x, NE);
  colstats<<<dim3(Cc), gb, 0, stream>>>(bufA, BNn, Cc, m1, i1);

  // 2) attention branch
  gemm_wmma<<<g256, gb, 0, stream>>>(xh, WqT, bq, nullptr, qh, BNn, Cc, Cc, 0);
  gemm_wmma<<<g256, gb, 0, stream>>>(xh, WkT, bk, nullptr, kh, BNn, Cc, Cc, 0);
  gemm_wmma<<<g256, gb, 0, stream>>>(xh, WvT, bv, nullptr, vh, BNn, Cc, Cc, 0);
  attn_wmma<<<dim3(Bn * Hh * (Nn / 64)), dim3(128), 0, stream>>>(qh, kh, vh,
                                                                 sph, oh);
  gemm_wmma<<<g256, gb, 0, stream>>>(oh, WoT, bo, bufC, nullptr, BNn, Cc, Cc, 0);
  add2_f32<<<eg, eb, 0, stream>>>(bufC, bufC, x, NE);
  colstats<<<dim3(Cc), gb, 0, stream>>>(bufC, BNn, Cc, m2, i2);

  // 3) s = bn1(t1) + bn2(t2); MLP residual; final BN
  bn2_sum<<<eg, eb, 0, stream>>>(bufA, bufC, m1, i1, g1, be1, m2, i2, g2, be2,
                                 sbuf, sh, NE);
  gemm_wmma<<<g512, gb, 0, stream>>>(sh, W1T, b1, nullptr, uh,
                                     BNn, 512, Cc, 1);
  gemm_wmma<<<g256, gb, 0, stream>>>(uh, W2T, b2, bufD, nullptr,
                                     BNn, Cc, 512, 0);
  add2_f32<<<eg, eb, 0, stream>>>(bufD, bufD, sbuf, NE);
  colstats<<<dim3(Cc), gb, 0, stream>>>(bufD, BNn, Cc, m3, i3);
  bn_apply<<<eg, eb, 0, stream>>>(bufD, m3, i3, g3, be3, out, NE);
}
